// InterpretableMultiHeadAttention_67138928771127
// MI455X (gfx1250) — compile-verified, matrix operand order checked by reading, not run on hardware
//
#include <hip/hip_runtime.h>

// ---------------------------------------------------------------------------
// InterpretableMultiHeadAttention for MI455X (gfx1250), bf16 WMMA pipeline.
// B=16, S=1024, HID=1024, NH=4, DH=256.
// Stages:
//   0) convert weights f32 -> bf16 (workspace)
//   1) q_s = q @ Wq[n]^T, k_s = k @ Wk[n]^T, v_s = v @ Wv^T   (bf16 WMMA)
//   2) scores = q_s @ k_s^T * 1/16 -> f32 straight into d_out attn region
//      in its final [B,S,NH,T] layout
//   3) softmax in place on d_out rows; bf16 copy of probs to workspace in
//      GEMM-friendly [B,NH,S,T] layout
//   4) head = attn @ v_s, fused mean over heads (accumulate 4 heads, *0.25)
//   5) outs = hmean @ Wh^T -> f32 into d_out
// All matmuls: v_wmma_f32_16x16x32_bf16. Block tile 128x128, K-step 32,
// 8 wave32 waves/block, each wave owns a 32x64 patch = 2x4 16x16 accumulator
// tiles (1.5 ds_load_b128 per WMMA, vs 2.0 with 32x32 tiles).
// ---------------------------------------------------------------------------

typedef __attribute__((ext_vector_type(16))) __bf16 v16bf;
typedef __attribute__((ext_vector_type(8)))  __bf16 bf16x8;
typedef __attribute__((ext_vector_type(8)))  float  v8f;

#define kB   16
#define kS   1024
#define kHID 1024
#define kNH  4
#define kDH  256

#define BM 128
#define BN 128
#define BK 32
#define LSTR 40          // LDS row stride (elements), padded vs 32 for banks
#define NTHR 256         // 8 wave32 waves

// ----------------------------- tile loaders --------------------------------

// 128x32 tile, f32 source row-major (lda), convert -> bf16 LDS (16B stores).
__device__ __forceinline__ void load_tile_f32(const float* __restrict__ A, int lda,
                                              __bf16* __restrict__ As, int tid) {
#pragma unroll
  for (int i = 0; i < 2; ++i) {
    int g = tid + i * NTHR;        // 0..511 chunks of 8
    int linear = g * 8;
    int row = linear >> 5;         // /32
    int kk  = linear & 31;
    float4 f0 = *(const float4*)(A + (size_t)row * lda + kk);
    float4 f1 = *(const float4*)(A + (size_t)row * lda + kk + 4);
    bf16x8 o;
    o[0] = (__bf16)f0.x; o[1] = (__bf16)f0.y; o[2] = (__bf16)f0.z; o[3] = (__bf16)f0.w;
    o[4] = (__bf16)f1.x; o[5] = (__bf16)f1.y; o[6] = (__bf16)f1.z; o[7] = (__bf16)f1.w;
    *(bf16x8*)(As + row * LSTR + kk) = o;
  }
}

// 128x32 tile, bf16 source row-major (lda) -> LDS (16B chunks).
// Serves both the A operand and the [N,K]-layout W operand (both 128 rows).
__device__ __forceinline__ void load_tile_bf16(const __bf16* __restrict__ A, int lda,
                                               __bf16* __restrict__ As, int tid) {
#pragma unroll
  for (int i = 0; i < 2; ++i) {
    int g = tid + i * NTHR;        // 0..511 chunks of 8
    int linear = g * 8;
    int row = linear >> 5;
    int kk  = linear & 31;
    *(bf16x8*)(As + row * LSTR + kk) = *(const bf16x8*)(A + (size_t)row * lda + kk);
  }
}

// V tile: source is [K=32, N=128] row-major (ldv); transpose into LDS [N][K].
__device__ __forceinline__ void load_tile_v_t(const __bf16* __restrict__ V, int ldv,
                                              __bf16* __restrict__ Ws, int tid) {
#pragma unroll
  for (int i = 0; i < 16; ++i) {
    int e = tid + i * NTHR;        // 0..4095, consecutive lanes -> consecutive n
    int k = e >> 7;                // /128
    int n = e & 127;
    Ws[n * LSTR + k] = V[(size_t)k * ldv + n];
  }
}

// ----------------------------- WMMA core -----------------------------------

__device__ __forceinline__ v16bf ld_frag(const __bf16* p) {
  union { v16bf v; bf16x8 h[2]; } u;
  u.h[0] = *(const bf16x8*)(p);
  u.h[1] = *(const bf16x8*)(p + 16);
  return u.v;
}

// One K-step (K=32): 2x4 16x16 output tiles per wave (wave patch 32x64).
__device__ __forceinline__ void mma_step(const __bf16* As, const __bf16* Ws,
                                         int wm, int wn, int lane, v8f (&acc)[2][4]) {
  const int lq = lane & 15;
  const int kb = (lane >> 4) * 8;  // half-wave K split per ISA fragment layout
  v16bf a[2], b[4];
#pragma unroll
  for (int i = 0; i < 2; ++i)
    a[i] = ld_frag(As + (wm * 32 + i * 16 + lq) * LSTR + kb);
#pragma unroll
  for (int j = 0; j < 4; ++j)
    b[j] = ld_frag(Ws + (wn * 64 + j * 16 + lq) * LSTR + kb);
#pragma unroll
  for (int i = 0; i < 2; ++i)
#pragma unroll
    for (int j = 0; j < 4; ++j)
      acc[i][j] = __builtin_amdgcn_wmma_f32_16x16x32_bf16(false, a[i], false, b[j],
                                                          (short)0, acc[i][j], false, false);
}

// C/D layout: lanes 0-15 hold M=0..7 (VGPR0..7) at N=lane; lanes 16-31 M=8..15.
__device__ __forceinline__ void store_acc_bf16(__bf16* C, int ldc, int m0, int n0,
                                               int wm, int wn, int lane,
                                               const v8f (&acc)[2][4], float scale) {
  const int lq = lane & 15;
  const int mb = (lane >> 4) * 8;
#pragma unroll
  for (int i = 0; i < 2; ++i)
#pragma unroll
    for (int j = 0; j < 4; ++j) {
      int row0 = m0 + wm * 32 + i * 16 + mb;
      int col  = n0 + wn * 64 + j * 16 + lq;
#pragma unroll
      for (int r = 0; r < 8; ++r)
        C[(size_t)(row0 + r) * ldc + col] = (__bf16)(acc[i][j][r] * scale);
    }
}

__device__ __forceinline__ void store_acc_f32(float* C, int ldc, int m0, int n0,
                                              int wm, int wn, int lane,
                                              const v8f (&acc)[2][4]) {
  const int lq = lane & 15;
  const int mb = (lane >> 4) * 8;
#pragma unroll
  for (int i = 0; i < 2; ++i)
#pragma unroll
    for (int j = 0; j < 4; ++j) {
      int row0 = m0 + wm * 32 + i * 16 + mb;
      int col  = n0 + wn * 64 + j * 16 + lq;
#pragma unroll
      for (int r = 0; r < 8; ++r)
        C[(size_t)(row0 + r) * ldc + col] = acc[i][j][r];
    }
}

#define ZERO8 {0,0,0,0,0,0,0,0}
#define ACC_INIT v8f acc[2][4] = { {ZERO8, ZERO8, ZERO8, ZERO8}, \
                                   {ZERO8, ZERO8, ZERO8, ZERO8} }

// ----------------------------- kernels -------------------------------------

__global__ void cvt_f32_bf16(const float* __restrict__ in, __bf16* __restrict__ out, int n) {
  int i = blockIdx.x * 256 + threadIdx.x;
  if (i < n) out[i] = (__bf16)in[i];
}

// Stage 1: per-head projection. X:[B,S,HID] f32, Wb:[nheads,DH,HID] bf16,
// Out:[B*nheads, S, DH] bf16. grid (DH/128, S/128, B*nheads).
__global__ __launch_bounds__(NTHR) void proj_kernel(const float* __restrict__ X,
                                                    const __bf16* __restrict__ Wb,
                                                    __bf16* __restrict__ Out, int nheads) {
  const int z = blockIdx.z;
  const int b = z / nheads, n = z % nheads;
  const int m0 = blockIdx.y * BM, n0 = blockIdx.x * BN;
  const int tid = threadIdx.x, lane = tid & 31, wave = tid >> 5;
  const int wm = wave >> 1, wn = wave & 1;
  __shared__ __align__(16) __bf16 As[BM * LSTR];
  __shared__ __align__(16) __bf16 Ws[BN * LSTR];
  ACC_INIT;
  const float* A = X + (size_t)b * kS * kHID + (size_t)m0 * kHID;
  const __bf16* W = Wb + (size_t)n * kDH * kHID + (size_t)n0 * kHID;
  for (int kk = 0; kk < kHID; kk += BK) {
    load_tile_f32(A + kk, kHID, As, tid);
    load_tile_bf16(W + kk, kHID, Ws, tid);
    if (kk + BK < kHID) {
      __builtin_prefetch(A + kk + BK + (size_t)(tid >> 1) * kHID, 0, 3);
      __builtin_prefetch(W + kk + BK + (size_t)(tid >> 1) * kHID, 0, 3);
    }
    __syncthreads();
    mma_step(As, Ws, wm, wn, lane, acc);
    __syncthreads();
  }
  store_acc_bf16(Out + (size_t)z * kS * kDH, kDH, m0, n0, wm, wn, lane, acc, 1.0f);
}

// Stage 2: scores = q_s @ k_s^T * 1/sqrt(DH). Writes f32 directly into d_out
// attn region in [B,S,NH,T] layout. grid (S/128, S/128, B*NH).
__global__ __launch_bounds__(NTHR) void scores_kernel(const __bf16* __restrict__ qs,
                                                      const __bf16* __restrict__ ks,
                                                      float* __restrict__ attn) {
  const int z = blockIdx.z;
  const int b = z >> 2, n = z & 3;
  const int m0 = blockIdx.y * BM, n0 = blockIdx.x * BN;
  const int tid = threadIdx.x, lane = tid & 31, wave = tid >> 5;
  const int wm = wave >> 1, wn = wave & 1;
  __shared__ __align__(16) __bf16 As[BM * LSTR];
  __shared__ __align__(16) __bf16 Ws[BN * LSTR];
  ACC_INIT;
  const __bf16* A = qs + (size_t)z * kS * kDH + (size_t)m0 * kDH;
  const __bf16* W = ks + (size_t)z * kS * kDH + (size_t)n0 * kDH;
  for (int kk = 0; kk < kDH; kk += BK) {
    load_tile_bf16(A + kk, kDH, As, tid);
    load_tile_bf16(W + kk, kDH, Ws, tid);
    __syncthreads();
    mma_step(As, Ws, wm, wn, lane, acc);
    __syncthreads();
  }
  // custom f32 store: element (s,t) -> attn[((b*S+s)*NH+n)*S + t], scaled 1/16
  const int lq = lane & 15;
  const int mb = (lane >> 4) * 8;
#pragma unroll
  for (int i = 0; i < 2; ++i)
#pragma unroll
    for (int j = 0; j < 4; ++j) {
      int s0 = m0 + wm * 32 + i * 16 + mb;
      int t  = n0 + wn * 64 + j * 16 + lq;
#pragma unroll
      for (int r = 0; r < 8; ++r)
        attn[((size_t)(b * kS + s0 + r) * kNH + n) * kS + t] = acc[i][j][r] * 0.0625f;
    }
}

// Stage 3: softmax over T, in place on f32 d_out rows; bf16 copy to workspace
// in [B,NH,S,T]. grid (B*S*NH) blocks of 256.
__global__ __launch_bounds__(NTHR) void softmax_kernel(float* __restrict__ attn,
                                                       __bf16* __restrict__ attnB) {
  const int r = blockIdx.x;              // r = (b*S+s)*NH + n
  const int tid = threadIdx.x;
  const int n = r & (kNH - 1);
  const int bs = r >> 2;
  const int b = bs >> 10;                // / S
  const int s = bs & (kS - 1);
  float* row = attn + (size_t)r * kS;
  __bf16* brow = attnB + ((size_t)(b * kNH + n) * kS + s) * kS;
  __shared__ float red[NTHR];
  float x[4];
  float m = -3.4e38f;
#pragma unroll
  for (int i = 0; i < 4; ++i) { x[i] = row[tid + i * NTHR]; m = fmaxf(m, x[i]); }
  red[tid] = m; __syncthreads();
  for (int off = NTHR / 2; off > 0; off >>= 1) {
    if (tid < off) red[tid] = fmaxf(red[tid], red[tid + off]);
    __syncthreads();
  }
  m = red[0]; __syncthreads();
  float sum = 0.0f;
#pragma unroll
  for (int i = 0; i < 4; ++i) { x[i] = __expf(x[i] - m); sum += x[i]; }
  red[tid] = sum; __syncthreads();
  for (int off = NTHR / 2; off > 0; off >>= 1) {
    if (tid < off) red[tid] += red[tid + off];
    __syncthreads();
  }
  float inv = 1.0f / red[0];
#pragma unroll
  for (int i = 0; i < 4; ++i) {
    float p = x[i] * inv;
    row[tid + i * NTHR] = p;
    brow[tid + i * NTHR] = (__bf16)p;
  }
}

// Stage 4: head = attn @ v_s with fused mean over heads (accumulate 4 heads,
// scale 0.25). grid (DH/128, S/128, B). Output bf16 [B,S,DH].
__global__ __launch_bounds__(NTHR) void head_gemm_kernel(const __bf16* __restrict__ attnB,
                                                         const __bf16* __restrict__ vs,
                                                         __bf16* __restrict__ hm) {
  const int b = blockIdx.z;
  const int m0 = blockIdx.y * BM, n0 = blockIdx.x * BN;
  const int tid = threadIdx.x, lane = tid & 31, wave = tid >> 5;
  const int wm = wave >> 1, wn = wave & 1;
  __shared__ __align__(16) __bf16 As[BM * LSTR];
  __shared__ __align__(16) __bf16 Ws[BN * LSTR];
  ACC_INIT;
  const __bf16* V = vs + (size_t)b * kS * kDH;
  for (int n = 0; n < kNH; ++n) {
    const __bf16* A = attnB + ((size_t)(b * kNH + n) * kS + m0) * kS;
    for (int kk = 0; kk < kS; kk += BK) {
      load_tile_bf16(A + kk, kS, As, tid);
      load_tile_v_t(V + (size_t)kk * kDH + n0, kDH, Ws, tid);
      if (kk + BK < kS)
        __builtin_prefetch(A + kk + BK + (size_t)(tid >> 1) * kS, 0, 3);
      __syncthreads();
      mma_step(As, Ws, wm, wn, lane, acc);
      __syncthreads();
    }
  }
  store_acc_bf16(hm + (size_t)b * kS * kDH, kDH, m0, n0, wm, wn, lane, acc, 0.25f);
}

// Stage 5: outs = hmean @ Wh^T, f32 out. grid (HID/128, S/128, B).
__global__ __launch_bounds__(NTHR) void out_gemm_kernel(const __bf16* __restrict__ hm,
                                                        const __bf16* __restrict__ Whb,
                                                        float* __restrict__ outs) {
  const int b = blockIdx.z;
  const int m0 = blockIdx.y * BM, n0 = blockIdx.x * BN;
  const int tid = threadIdx.x, lane = tid & 31, wave = tid >> 5;
  const int wm = wave >> 1, wn = wave & 1;
  __shared__ __align__(16) __bf16 As[BM * LSTR];
  __shared__ __align__(16) __bf16 Ws[BN * LSTR];
  ACC_INIT;
  const __bf16* A = hm + (size_t)b * kS * kDH + (size_t)m0 * kDH;
  const __bf16* W = Whb + (size_t)n0 * kDH;
  for (int kk = 0; kk < kDH; kk += BK) {
    load_tile_bf16(A + kk, kDH, As, tid);
    load_tile_bf16(W + kk, kDH, Ws, tid);
    __syncthreads();
    mma_step(As, Ws, wm, wn, lane, acc);
    __syncthreads();
  }
  store_acc_f32(outs + (size_t)b * kS * kHID, kHID, m0, n0, wm, wn, lane, acc);
}

// ----------------------------- launcher ------------------------------------

extern "C" void kernel_launch(void* const* d_in, const int* in_sizes, int n_in,
                              void* d_out, int out_size, void* d_ws, size_t ws_size,
                              hipStream_t stream) {
  (void)in_sizes; (void)n_in; (void)out_size; (void)ws_size;
  const float* q  = (const float*)d_in[0];
  const float* k  = (const float*)d_in[1];
  const float* v  = (const float*)d_in[2];
  const float* Wq = (const float*)d_in[3];
  const float* Wk = (const float*)d_in[4];
  const float* Wv = (const float*)d_in[5];
  const float* Wh = (const float*)d_in[6];

  float* outs = (float*)d_out;                          // [B,S,HID]
  float* attn = outs + (size_t)kB * kS * kHID;          // [B,S,NH,T]

  // workspace carve-up (~222 MB total)
  char* wsb = (char*)d_ws;
  size_t off = 0;
  auto carve = [&](size_t bytes) -> char* {
    char* p = wsb + off;
    off = (off + bytes + 255) & ~(size_t)255;
    return p;
  };
  __bf16* Wq_b  = (__bf16*)carve((size_t)kNH * kDH * kHID * 2);     // 2 MB
  __bf16* Wk_b  = (__bf16*)carve((size_t)kNH * kDH * kHID * 2);     // 2 MB
  __bf16* Wv_b  = (__bf16*)carve((size_t)kDH * kHID * 2);           // 0.5 MB
  __bf16* Wh_b  = (__bf16*)carve((size_t)kHID * kDH * 2);           // 0.5 MB
  __bf16* q_s   = (__bf16*)carve((size_t)kB * kNH * kS * kDH * 2);  // 33.5 MB
  __bf16* k_s   = (__bf16*)carve((size_t)kB * kNH * kS * kDH * 2);  // 33.5 MB
  __bf16* v_s   = (__bf16*)carve((size_t)kB * kS * kDH * 2);        // 8.4 MB
  __bf16* attnB = (__bf16*)carve((size_t)kB * kNH * kS * kS * 2);   // 134 MB
  __bf16* hmean = (__bf16*)carve((size_t)kB * kS * kDH * 2);        // 8.4 MB

  // Stage 0: weight conversion
  {
    int n1 = kNH * kDH * kHID;   // 1,048,576
    int n2 = kDH * kHID;         // 262,144
    cvt_f32_bf16<<<dim3((n1 + 255) / 256), dim3(256), 0, stream>>>(Wq, Wq_b, n1);
    cvt_f32_bf16<<<dim3((n1 + 255) / 256), dim3(256), 0, stream>>>(Wk, Wk_b, n1);
    cvt_f32_bf16<<<dim3((n2 + 255) / 256), dim3(256), 0, stream>>>(Wv, Wv_b, n2);
    cvt_f32_bf16<<<dim3((n2 + 255) / 256), dim3(256), 0, stream>>>(Wh, Wh_b, n2);
  }

  const dim3 blk(NTHR);
  // Stage 1: projections
  proj_kernel<<<dim3(kDH / BN, kS / BM, kB * kNH), blk, 0, stream>>>(q, Wq_b, q_s, kNH);
  proj_kernel<<<dim3(kDH / BN, kS / BM, kB * kNH), blk, 0, stream>>>(k, Wk_b, k_s, kNH);
  proj_kernel<<<dim3(kDH / BN, kS / BM, kB),       blk, 0, stream>>>(v, Wv_b, v_s, 1);
  // Stage 2: scores -> f32 into d_out attn region
  scores_kernel<<<dim3(kS / BN, kS / BM, kB * kNH), blk, 0, stream>>>(q_s, k_s, attn);
  // Stage 3: softmax in place + bf16 copy
  softmax_kernel<<<dim3(kB * kS * kNH), blk, 0, stream>>>(attn, attnB);
  // Stage 4: attn @ v_s, fused head-mean
  head_gemm_kernel<<<dim3(kDH / BN, kS / BM, kB), blk, 0, stream>>>(attnB, v_s, hmean);
  // Stage 5: output projection
  out_gemm_kernel<<<dim3(kHID / BN, kS / BM, kB), blk, 0, stream>>>(hmean, Wh_b, outs);
}
